// Head_13666585935906
// MI455X (gfx1250) — matrix operand order repaired and checked by reading, not compiled, not run on hardware
//
#include <hip/hip_runtime.h>
#include <hip/hip_bf16.h>

typedef __attribute__((ext_vector_type(16))) __bf16 v16bf;
typedef __attribute__((ext_vector_type(8)))  __bf16 v8bf;
typedef __attribute__((ext_vector_type(8)))  float  v8f;
typedef __attribute__((ext_vector_type(4)))  float  v4f;
typedef __attribute__((ext_vector_type(4)))  int    v4i;

#define B_  512
#define T_  256
#define C_  384
#define H_  64

// workspace layout (bytes)
#define QS_OFF   ((size_t)256*1024)
#define KS_OFF   (QS_OFF + (size_t)B_*T_*H_*2)
#define VT_OFF   (KS_OFF + (size_t)B_*T_*H_*2)

// CDNA5 async global -> LDS copy (16 bytes per lane, tracked by ASYNCcnt).
// Builtin prototype (per hipcc diagnostic): (v4i AS(1)*, v4i AS(3)*, imm, imm)
__device__ __forceinline__ void async_cp16(void* lds, const void* glob) {
    __builtin_amdgcn_global_load_async_to_lds_b128(
        (__attribute__((address_space(1))) v4i*)(v4i*)glob,
        (__attribute__((address_space(3))) v4i*)(v4i*)lds,
        0, 0);
}

// ---------------------------------------------------------------------------
// Kernel 0: W (C x H, fp32, row-major) -> Wt (3 x H x C, bf16)  [q, k, v]
// ---------------------------------------------------------------------------
__global__ void wconv_kernel(const float* __restrict__ Wq,
                             const float* __restrict__ Wk,
                             const float* __restrict__ Wv,
                             __bf16* __restrict__ Wt) {
    int idx = blockIdx.x * 256 + threadIdx.x;
    if (idx >= 3 * C_ * H_) return;
    int mat = idx / (C_ * H_);
    int rem = idx - mat * (C_ * H_);
    int c = rem / H_;
    int h = rem - c * H_;
    const float* W = (mat == 0) ? Wq : ((mat == 1) ? Wk : Wv);
    Wt[(size_t)mat * H_ * C_ + (size_t)h * C_ + c] = (__bf16)W[(size_t)c * H_ + h];
}

// ---------------------------------------------------------------------------
// Kernel 1: projections.  One wave per 16-row tile of x (M = B*T = 131072).
// Wt (147456 B bf16) is async-staged to LDS once per workgroup; the inner
// loop feeds B fragments from LDS (ds_load) instead of vmem.
// Outputs: Qs (scaled, [B*T,H] bf16), Ks ([B*T,H] bf16), Vt ([B,H,T] bf16).
// ---------------------------------------------------------------------------
__global__ void __launch_bounds__(256) proj_kernel(const float* __restrict__ x,
                                                   const __bf16* __restrict__ Wt,
                                                   __bf16* __restrict__ Qs,
                                                   __bf16* __restrict__ Ks,
                                                   __bf16* __restrict__ Vt) {
    extern __shared__ char smem[];                 // 147456 B: Wt staged
    const int lane = threadIdx.x & 31;
    const int wave = threadIdx.x >> 5;
    const int tile = blockIdx.x * 8 + wave;
    const int gt0  = tile * 16;
    const int m    = lane & 15;
    const bool hi  = lane >= 16;
    const float* xr = x + (size_t)(gt0 + m) * C_;

    // cooperative async stage of all three weight matrices into LDS
    {
        const char* gw = (const char*)Wt;
#pragma unroll
        for (int i = 0; i < 36; i++) {             // 9216 chunks of 16B / 256 thr
            const int off = (threadIdx.x + i * 256) * 16;
            async_cp16(smem + off, gw + off);
        }
        asm volatile("s_wait_asynccnt 0" ::: "memory");
        __syncthreads();
    }
    const __bf16* Wl = (const __bf16*)smem;

    v8f acc[12];
#pragma unroll
    for (int i = 0; i < 12; i++) acc[i] = (v8f)0.0f;

    for (int kk = 0; kk < 12; kk++) {
        const int c0 = kk * 32;
        const int ca = c0 + (hi ? 16 : 0);
        // A fragment: 16x32 bf16, contiguous k = 16*half + s per lane
        v4f a0 = *(const v4f*)(xr + ca);
        v4f a1 = *(const v4f*)(xr + ca + 4);
        v4f a2 = *(const v4f*)(xr + ca + 8);
        v4f a3 = *(const v4f*)(xr + ca + 12);
        v16bf A;
#pragma unroll
        for (int j = 0; j < 4; j++) {
            A[j]      = (__bf16)a0[j];
            A[4 + j]  = (__bf16)a1[j];
            A[8 + j]  = (__bf16)a2[j];
            A[12 + j] = (__bf16)a3[j];
        }
        const __bf16* wb = Wl + (size_t)(c0 + (hi ? 16 : 0));
#pragma unroll
        for (int mat = 0; mat < 3; mat++) {
#pragma unroll
            for (int nt = 0; nt < 4; nt++) {
                const __bf16* bp = wb + ((size_t)mat * H_ + nt * 16 + m) * C_;
                v16bf Bf = *(const v16bf*)bp;
                acc[mat * 4 + nt] = __builtin_amdgcn_wmma_f32_16x16x32_bf16(
                    false, A, false, Bf, (short)0, acc[mat * 4 + nt], false, false);
            }
        }
    }

    const float qscale = 0.051031036307982884f;  // 384^-0.5 (reference scales by C)
    const int b  = gt0 >> 8;
    const int tb = gt0 & 255;
#pragma unroll
    for (int nt = 0; nt < 4; nt++) {
        const int h = nt * 16 + m;
#pragma unroll
        for (int r = 0; r < 8; r++) {
            const int row = r + (hi ? 8 : 0);
            const size_t g = (size_t)(gt0 + row) * H_ + h;
            Qs[g] = (__bf16)(acc[nt][r] * qscale);
            Ks[g] = (__bf16)(acc[4 + nt][r]);
            Vt[(size_t)b * H_ * T_ + (size_t)h * T_ + (tb + row)] = (__bf16)(acc[8 + nt][r]);
        }
    }
}

// ---------------------------------------------------------------------------
// Kernel 2: causal flash attention.  8 waves/block, all on the same batch:
// K (32KB) + Vt (32KB) async-staged to LDS once, 8x reuse from LDS.
// Per 32-key step: 4 WMMA scores + online softmax + LDS C->A relayout + 4 WMMA PV.
// ---------------------------------------------------------------------------
__global__ void __launch_bounds__(256) attn_kernel(const __bf16* __restrict__ Qs,
                                                   const __bf16* __restrict__ Ks,
                                                   const __bf16* __restrict__ Vt,
                                                   float* __restrict__ out) {
    extern __shared__ char smem[];                 // 32K K + 32K Vt + 10K P
    __bf16* Klds = (__bf16*)smem;                  // [256][64]
    __bf16* Vlds = (__bf16*)(smem + 32768);        // [64][256]
    __bf16* Plds = (__bf16*)(smem + 65536);        // [8][16][40]

    const int lane = threadIdx.x & 31;
    const int wave = threadIdx.x >> 5;
    const int b  = blockIdx.x >> 1;
    const int qt = (blockIdx.x & 1) * 8 + wave;
    const int q0 = qt * 16;
    const int m  = lane & 15;
    const bool hi = lane >= 16;

    // cooperative async stage of this batch's K and Vt into LDS
    {
        const char* gk = (const char*)(Ks + (size_t)b * T_ * H_);
        const char* gv = (const char*)(Vt + (size_t)b * H_ * T_);
#pragma unroll
        for (int i = 0; i < 8; i++) {              // 2048 chunks of 16B / 256 thr
            const int off = (threadIdx.x + i * 256) * 16;
            async_cp16(smem + off, gk + off);
            async_cp16(smem + 32768 + off, gv + off);
        }
        asm volatile("s_wait_asynccnt 0" ::: "memory");
        __syncthreads();
    }

    // Q A-fragments (16x32 each, covering head dims [0,32) and [32,64))
    const __bf16* qr = Qs + ((size_t)(b * T_ + q0 + m)) * H_;
    v8bf qa = *(const v8bf*)(qr + (hi ? 16 : 0));
    v8bf qb = *(const v8bf*)(qr + 8 + (hi ? 16 : 0));
    v8bf qc = *(const v8bf*)(qr + 32 + (hi ? 16 : 0));
    v8bf qd = *(const v8bf*)(qr + 40 + (hi ? 16 : 0));
    v16bf Qa0, Qa1;
#pragma unroll
    for (int j = 0; j < 8; j++) { Qa0[j] = qa[j]; Qa0[8 + j] = qb[j]; Qa1[j] = qc[j]; Qa1[8 + j] = qd[j]; }

    v8f o[4];
#pragma unroll
    for (int i = 0; i < 4; i++) o[i] = (v8f)0.0f;
    float mrun[8], lrun[8];
#pragma unroll
    for (int r = 0; r < 8; r++) { mrun[r] = -1e30f; lrun[r] = 0.0f; }

    const int nsteps = (qt >> 1) + 1;              // ceil((qt+1)*16 / 32)

    for (int s = 0; s < nsteps; s++) {
        const int kb = s * 32;
        v8f s0 = (v8f)0.0f, s1 = (v8f)0.0f;
        {
            const __bf16* kr0 = Klds + (size_t)(kb + m) * H_ + (hi ? 16 : 0);
            v16bf K0a = *(const v16bf*)kr0;
            v16bf K0b = *(const v16bf*)(kr0 + 32);
            s0 = __builtin_amdgcn_wmma_f32_16x16x32_bf16(false, Qa0, false, K0a, (short)0, s0, false, false);
            s0 = __builtin_amdgcn_wmma_f32_16x16x32_bf16(false, Qa1, false, K0b, (short)0, s0, false, false);
            const __bf16* kr1 = Klds + (size_t)(kb + 16 + m) * H_ + (hi ? 16 : 0);
            v16bf K1a = *(const v16bf*)kr1;
            v16bf K1b = *(const v16bf*)(kr1 + 32);
            s1 = __builtin_amdgcn_wmma_f32_16x16x32_bf16(false, Qa0, false, K1a, (short)0, s1, false, false);
            s1 = __builtin_amdgcn_wmma_f32_16x16x32_bf16(false, Qa1, false, K1b, (short)0, s1, false, false);
        }
        // causal mask + online softmax (rows live on 16-lane halves -> xor 1,2,4,8)
        const int k0 = kb + m;
        const int k1 = kb + 16 + m;
        const int qrow0 = q0 + (hi ? 8 : 0);
#pragma unroll
        for (int r = 0; r < 8; r++) {
            const int qi = qrow0 + r;
            float e0 = (k0 <= qi) ? s0[r] : -1e30f;
            float e1 = (k1 <= qi) ? s1[r] : -1e30f;
            float mx = fmaxf(e0, e1);
            mx = fmaxf(mx, __shfl_xor(mx, 1));
            mx = fmaxf(mx, __shfl_xor(mx, 2));
            mx = fmaxf(mx, __shfl_xor(mx, 4));
            mx = fmaxf(mx, __shfl_xor(mx, 8));
            float mn = fmaxf(mrun[r], mx);
            float al = __expf(mrun[r] - mn);
            mrun[r] = mn;
            float p0 = __expf(e0 - mn);
            float p1 = __expf(e1 - mn);
            float rs = p0 + p1;
            rs += __shfl_xor(rs, 1);
            rs += __shfl_xor(rs, 2);
            rs += __shfl_xor(rs, 4);
            rs += __shfl_xor(rs, 8);
            lrun[r] = lrun[r] * al + rs;
#pragma unroll
            for (int nt = 0; nt < 4; nt++) o[nt][r] *= al;
            const int prow = r + (hi ? 8 : 0);
            Plds[(size_t)wave * 640 + prow * 40 + m]      = (__bf16)p0;
            Plds[(size_t)wave * 640 + prow * 40 + 16 + m] = (__bf16)p1;
        }
        // intra-wave LDS C-layout -> A-layout relayout: DS is in-order per wave,
        // the wait + memory clobber stop the compiler hoisting the reads.
        asm volatile("s_wait_dscnt 0" ::: "memory");
        v16bf Pa;
        {
            const __bf16* pr = Plds + (size_t)wave * 640 + m * 40;
            v8bf pa0 = *(const v8bf*)(pr + (hi ? 16 : 0));
            v8bf pa1 = *(const v8bf*)(pr + 8 + (hi ? 16 : 0));
#pragma unroll
            for (int j = 0; j < 8; j++) { Pa[j] = pa0[j]; Pa[8 + j] = pa1[j]; }
        }
        asm volatile("" ::: "memory");
#pragma unroll
        for (int nt = 0; nt < 4; nt++) {
            const __bf16* vp = Vlds + (size_t)(nt * 16 + m) * T_ + kb + (hi ? 16 : 0);
            v16bf Vb = *(const v16bf*)vp;
            o[nt] = __builtin_amdgcn_wmma_f32_16x16x32_bf16(false, Pa, false, Vb, (short)0, o[nt], false, false);
        }
    }

#pragma unroll
    for (int r = 0; r < 8; r++) {
        const float inv = 1.0f / lrun[r];
        const int row = q0 + r + (hi ? 8 : 0);
#pragma unroll
        for (int nt = 0; nt < 4; nt++) {
            out[((size_t)(b * T_ + row)) * H_ + nt * 16 + m] = o[nt][r] * inv;
        }
    }
}

// ---------------------------------------------------------------------------
extern "C" void kernel_launch(void* const* d_in, const int* in_sizes, int n_in,
                              void* d_out, int out_size, void* d_ws, size_t ws_size,
                              hipStream_t stream) {
    (void)in_sizes; (void)n_in; (void)out_size; (void)ws_size;
    const float* x  = (const float*)d_in[0];
    const float* Wk = (const float*)d_in[1];
    const float* Wq = (const float*)d_in[2];
    const float* Wv = (const float*)d_in[3];
    char* ws = (char*)d_ws;
    __bf16* Wt = (__bf16*)ws;
    __bf16* Qs = (__bf16*)(ws + QS_OFF);
    __bf16* Ks = (__bf16*)(ws + KS_OFF);
    __bf16* Vt = (__bf16*)(ws + VT_OFF);
    float* out = (float*)d_out;

    wconv_kernel<<<dim3((3 * C_ * H_ + 255) / 256), dim3(256), 0, stream>>>(Wq, Wk, Wv, Wt);
    proj_kernel<<<dim3((B_ * T_ / 16) / 8), dim3(256), 147456, stream>>>(x, Wt, Qs, Ks, Vt);
    attn_kernel<<<dim3(B_ * (T_ / 16) / 8), dim3(256), 75776, stream>>>(Qs, Ks, Vt, out);
}
